// GATModel_46995532152905
// MI455X (gfx1250) — matrix operand order repaired and checked by reading, not compiled, not run on hardware
//
#include <hip/hip_runtime.h>

#define NND 10000
#define NED 160000

typedef _Float16 v16h __attribute__((ext_vector_type(16)));
typedef float    v8f  __attribute__((ext_vector_type(8)));

// ---- ordered-uint encoding for float atomic max ----
__device__ __forceinline__ unsigned encf(float f) {
    unsigned u = __float_as_uint(f);
    return (u & 0x80000000u) ? ~u : (u | 0x80000000u);
}
__device__ __forceinline__ float decf(unsigned e) {
    unsigned u = (e & 0x80000000u) ? (e & 0x7FFFFFFFu) : ~e;
    return __uint_as_float(u);
}

// ---- conversions ----
__global__ void k_f32_to_f16(const float* __restrict__ in, _Float16* __restrict__ out, int n) {
    int i = blockIdx.x * blockDim.x + threadIdx.x;
    if (i < n) out[i] = (_Float16)in[i];
}

// W [cin, nout] f32 row-major  ->  Wt [nout, cin] f16 row-major
__global__ void k_w_to_f16_t(const float* __restrict__ W, _Float16* __restrict__ Wt,
                             int cin, int nout) {
    int i = blockIdx.x * blockDim.x + threadIdx.x;
    if (i < cin * nout) {
        int n = i / cin, k = i - n * cin;
        Wt[i] = (_Float16)W[(size_t)k * nout + n];
    }
}

__global__ void k_fill_f32(float* __restrict__ p, float v, int n) {
    int i = blockIdx.x * blockDim.x + threadIdx.x;
    if (i < n) p[i] = v;
}
__global__ void k_fill_u32(unsigned* __restrict__ p, unsigned v, int n) {
    int i = blockIdx.x * blockDim.x + threadIdx.x;
    if (i < n) p[i] = v;
}

// ---- WMMA GEMM: C[M,Nc] = A[M,K] * Bt[Nc,K]^T, f16 in, f32 out ----
// Each wave computes a 16x64 strip (4 WMMA n-tiles sharing one A fragment).
// 8 waves per 256-thread block.  No predicated loads: out-of-range rows are
// clamped (their products land only in discarded output rows/cols).
// Requires M % 16 == 0 (true here: M = 10000) and K % 32 == 0.
__global__ __launch_bounds__(256) void k_gemm_f16_wmma(
    const _Float16* __restrict__ A,   // [M, K]   row-major
    const _Float16* __restrict__ Bt,  // [Nc, K]  row-major (transposed weights)
    float* __restrict__ C,            // [M, Nc]
    int M, int K, int Nc, int nQTiles, int totTiles)
{
    int wave = threadIdx.x >> 5;
    int lane = threadIdx.x & 31;
    int tile = blockIdx.x * 8 + wave;
    if (tile >= totTiles) return;
    int mT = tile / nQTiles, nQ = tile - mT * nQTiles;
    int half = lane >> 4;       // lane-half selects K sub-range
    int lm   = lane & 15;       // A-row / B-col within tile

    // A fragment: lane lm = row, per-half K split; two contiguous 16B runs:
    //   K = half*8 + [0..7]  and  K = 16 + half*8 + [0..7]
    const _Float16* aRow = A + (size_t)(mT * 16 + lm) * K + half * 8;

    // B fragment: lane lm = column, per-half K split; two contiguous 16B runs:
    //   K = half*8 + [0..7]  and  K = 16 + half*8 + [0..7]
    const _Float16* bRow[4];
#pragma unroll
    for (int q = 0; q < 4; ++q) {
        int rb = nQ * 64 + q * 16 + lm;
        if (rb >= Nc) rb = Nc - 1;        // clamp; those columns never stored
        bRow[q] = Bt + (size_t)rb * K + half * 8;
    }

    union Frag { v16h v; uint4 q[2]; };
    v8f acc[4] = {v8f{}, v8f{}, v8f{}, v8f{}};

    for (int k = 0; k < K; k += 32) {
        Frag a;
        a.q[0] = *(const uint4*)(aRow + k);
        a.q[1] = *(const uint4*)(aRow + k + 16);
        Frag b[4];
#pragma unroll
        for (int q = 0; q < 4; ++q) {
            b[q].q[0] = *(const uint4*)(bRow[q] + k);
            b[q].q[1] = *(const uint4*)(bRow[q] + k + 16);
        }
        if (k + 32 < K) {   // gfx1250 global_prefetch_b8 for next K-slab
            __builtin_prefetch(aRow + k + 32, 0, 1);
            __builtin_prefetch(bRow[0] + k + 32, 0, 1);
        }
#pragma unroll
        for (int q = 0; q < 4; ++q)
            acc[q] = __builtin_amdgcn_wmma_f32_16x16x32_f16(
                /*neg_a=*/false, a.v, /*neg_b=*/false, b[q].v,
                /*c_mod=*/(short)0, acc[q], /*reuse_a=*/false, /*reuse_b=*/false);
    }

    // D layout: lane half picks M base (0/8), VGPR r -> row, col = lane&15
    int mBase = mT * 16 + half * 8;
#pragma unroll
    for (int q = 0; q < 4; ++q) {
        int col = nQ * 64 + q * 16 + lm;
        if (col < Nc) {
#pragma unroll
            for (int r = 0; r < 8; ++r)
                C[(size_t)(mBase + r) * Nc + col] = acc[q][r];
        }
    }
}

// ---- per-node attention logits: als/ald[n,h] = sum_c h[n,h,c]*a[h,c] ----
__global__ void k_attn_logits(const float* __restrict__ h, const float* __restrict__ asrc,
                              const float* __restrict__ adst, float* __restrict__ als,
                              float* __restrict__ ald, int N, int H, int C) {
    int i = blockIdx.x * blockDim.x + threadIdx.x;
    if (i >= N * H) return;
    int n = i / H, hh = i - n * H;
    const float* hr = h + (size_t)n * H * C + (size_t)hh * C;
    const float* as = asrc + hh * C;
    const float* ad = adst + hh * C;
    float s0 = 0.f, s1 = 0.f;
    for (int c = 0; c < C; ++c) { float v = hr[c]; s0 += v * as[c]; s1 += v * ad[c]; }
    als[i] = s0; ald[i] = s1;
}

// ---- edge pass 1: leaky-relu logit + segment max over dst ----
__global__ void k_edge_logit_max(const int* __restrict__ ei, const float* __restrict__ als,
                                 const float* __restrict__ ald, float* __restrict__ ebuf,
                                 unsigned* __restrict__ maxe, int E, int H) {
    int i = blockIdx.x * blockDim.x + threadIdx.x;
    if (i >= E * H) return;
    int e = i / H, hh = i - e * H;
    int s = ei[e], d = ei[E + e];
    float v = als[s * H + hh] + ald[d * H + hh];
    v = (v > 0.f) ? v : 0.2f * v;          // leaky relu, slope 0.2
    ebuf[i] = v;
    atomicMax(&maxe[d * H + hh], encf(v));
}

// ---- edge pass 2: exp(e - max) + segment sum ----
__global__ void k_edge_exp_sum(const int* __restrict__ ei, float* __restrict__ ebuf,
                               const unsigned* __restrict__ maxe, float* __restrict__ den,
                               int E, int H) {
    int i = blockIdx.x * blockDim.x + threadIdx.x;
    if (i >= E * H) return;
    int e = i / H, hh = i - e * H;
    int d = ei[E + e];
    float w = expf(ebuf[i] - decf(maxe[d * H + hh]));
    ebuf[i] = w;
    atomicAdd(&den[d * H + hh], w);
}

// ---- edge pass 3: out[dst] += alpha * h[src] ----
__global__ void k_aggregate(const int* __restrict__ ei, const float* __restrict__ ebuf,
                            const float* __restrict__ den, const float* __restrict__ h,
                            float* __restrict__ out, int E, int H, int C) {
    int e = blockIdx.x;
    int s = ei[e], d = ei[E + e];
    int HC = H * C;
    for (int idx = threadIdx.x; idx < HC; idx += blockDim.x) {
        int hh = idx / C;
        float alpha = ebuf[(size_t)e * H + hh] / (den[d * H + hh] + 1e-16f);
        atomicAdd(&out[(size_t)d * HC + idx], h[(size_t)s * HC + idx] * alpha);
    }
}

// ---- bias + optional relu ----
__global__ void k_bias_act(const float* __restrict__ in, const float* __restrict__ b,
                           float* __restrict__ out, int N, int HC, int doRelu) {
    int i = blockIdx.x * blockDim.x + threadIdx.x;
    if (i >= N * HC) return;
    float v = in[i] + b[i % HC];
    if (doRelu) v = v > 0.f ? v : 0.f;
    out[i] = v;
}

extern "C" void kernel_launch(void* const* d_in, const int* in_sizes, int n_in,
                              void* d_out, int out_size, void* d_ws, size_t ws_size,
                              hipStream_t stream) {
    (void)in_sizes; (void)n_in; (void)out_size; (void)ws_size;
    const int N = NND, E = NED;
    const float* x  = (const float*)d_in[0];
    const int*   ei = (const int*)d_in[1];   // [2, E]: row0 = src, row1 = dst

    char* ws = (char*)d_ws;
    size_t off = 0;
    auto wsalloc = [&](size_t bytes) -> void* {
        void* p = ws + off; off += (bytes + 255) & ~(size_t)255; return p;
    };
    float*    B0   = (float*)   wsalloc((size_t)N * 1024 * 4);
    float*    B1   = (float*)   wsalloc((size_t)N * 1024 * 4);
    float*    B2   = (float*)   wsalloc((size_t)N * 512  * 4);
    _Float16* ACT  = (_Float16*)wsalloc((size_t)N * 1024 * 2);
    _Float16* WT   = (_Float16*)wsalloc((size_t)1024 * 512 * 2);
    float*    ALS  = (float*)   wsalloc((size_t)N * 8 * 4);
    float*    ALD  = (float*)   wsalloc((size_t)N * 8 * 4);
    unsigned* MAXE = (unsigned*)wsalloc((size_t)N * 8 * 4);
    float*    DEN  = (float*)   wsalloc((size_t)N * 8 * 4);
    float*    EBUF = (float*)   wsalloc((size_t)E * 8 * 4);

    struct Cfg { int cin, cout, H, wIdx; const float* in; float* h; float* out; float* actdst; int relu; };
    Cfg cfg[4] = {
        { 256, 128, 8,  2, x,  B0, B1, B1,            1},
        {1024,  64, 8,  6, B1, B0, B2, B2,            1},
        { 512,  32, 8, 10, B2, B1, B0, B0,            1},
        { 256,   1, 1, 14, B0, B1, B2, (float*)d_out, 0},
    };

    const int T = 256;
    for (int i = 0; i < 4; ++i) {
        Cfg& L = cfg[i];
        int HC = L.H * L.cout;
        const float* W    = (const float*)d_in[L.wIdx];
        const float* asrc = (const float*)d_in[L.wIdx + 1];
        const float* adst = (const float*)d_in[L.wIdx + 2];
        const float* bias = (const float*)d_in[L.wIdx + 3];

        int nAct = N * L.cin;
        k_f32_to_f16<<<(nAct + T - 1) / T, T, 0, stream>>>(L.in, ACT, nAct);
        int nW = L.cin * HC;
        k_w_to_f16_t<<<(nW + T - 1) / T, T, 0, stream>>>(W, WT, L.cin, HC);

        int mTiles = (N + 15) / 16;            // 625 (exact)
        int nQTiles = (HC + 63) / 64;          // 16 / 8 / 4 / 1
        int tot = mTiles * nQTiles;
        k_gemm_f16_wmma<<<(tot + 7) / 8, 256, 0, stream>>>(ACT, WT, L.h, N, L.cin, HC, nQTiles, tot);

        int nNH = N * L.H;
        k_attn_logits<<<(nNH + T - 1) / T, T, 0, stream>>>(L.h, asrc, adst, ALS, ALD, N, L.H, L.cout);

        k_fill_u32<<<(nNH + T - 1) / T, T, 0, stream>>>(MAXE, 0x007FFFFFu /*enc(-inf)*/, nNH);
        k_fill_f32<<<(nNH + T - 1) / T, T, 0, stream>>>(DEN, 0.f, nNH);
        int nOut = N * HC;
        k_fill_f32<<<(nOut + T - 1) / T, T, 0, stream>>>(L.out, 0.f, nOut);

        int nEH = E * L.H;
        k_edge_logit_max<<<(nEH + T - 1) / T, T, 0, stream>>>(ei, ALS, ALD, EBUF, MAXE, E, L.H);
        k_edge_exp_sum <<<(nEH + T - 1) / T, T, 0, stream>>>(ei, EBUF, MAXE, DEN, E, L.H);
        k_aggregate    <<<E, 256, 0, stream>>>(ei, EBUF, DEN, L.h, L.out, E, L.H, L.cout);

        k_bias_act<<<(nOut + T - 1) / T, T, 0, stream>>>(L.out, bias, L.actdst, N, HC, L.relu);
    }
}
